// DSTAGNN_submodule_68169720922839
// MI455X (gfx1250) — matrix operand order repaired and checked by reading, not compiled, not run on hardware
//
#include <hip/hip_runtime.h>
#include <hip/hip_bf16.h>
#include <math.h>

typedef __attribute__((ext_vector_type(16))) _Float16 v16h;
typedef __attribute__((ext_vector_type(8)))  float    v8f;

#define B_    32
#define N_    512
#define T_    12
#define H_    3
#define K_    3
#define FOUT_ 32

// ---------------------------------------------------------------------------
// Block reductions (blockDim.x == 256, power of two)
// ---------------------------------------------------------------------------
__device__ __forceinline__ float blockReduceSum(float v, float* sh) {
  int tid = threadIdx.x;
  sh[tid] = v; __syncthreads();
  for (int s = 128; s > 0; s >>= 1) {
    if (tid < s) sh[tid] += sh[tid + s];
    __syncthreads();
  }
  float r = sh[0]; __syncthreads();
  return r;
}
__device__ __forceinline__ float blockReduceMax(float v, float* sh) {
  int tid = threadIdx.x;
  sh[tid] = v; __syncthreads();
  for (int s = 128; s > 0; s >>= 1) {
    if (tid < s) sh[tid] = fmaxf(sh[tid], sh[tid + s]);
    __syncthreads();
  }
  float r = sh[0]; __syncthreads();
  return r;
}

// ---------------------------------------------------------------------------
// Fragment loaders: fp32 gather -> fp16 fragment in the CDNA5 WMMA layouts.
//  KG : K-tail guard (clamp + zero-select, branch-free)
//  AK1: A is k-contiguous (sa_k==1, no K guard) -> float2 (b64) loads
//  BK1: B is k-contiguous (sb_k==1, no guards)  -> float4 (b128) loads
// ---------------------------------------------------------------------------
template<int AMODE, int P1, bool KG, bool AK1>
__device__ __forceinline__ v16h loadAfrag(const float* __restrict__ Ap, long rowBase,
                                          bool rowOK, int kk, int hiA, long sa_k, int Kd)
{
  float f[16];
  if (AK1) {
#pragma unroll
    for (int v = 0; v < 8; ++v) {
      const int kb = kk + hiA + 2 * v;
      const float2 p = *(const float2*)(Ap + rowBase + kb);   // 8B aligned
      f[2 * v]     = p.x;
      f[2 * v + 1] = p.y;
    }
  } else {
#pragma unroll
    for (int v = 0; v < 8; ++v) {
      const int kb = kk + hiA + 2 * v;
#pragma unroll
      for (int e = 0; e < 2; ++e) {
        const int k  = kb + e;
        const int kc = KG ? ((k < Kd) ? k : 0) : k;
        long idx;
        if (AMODE == 0) idx = rowBase + (long)kc * sa_k;
        else            idx = rowBase + (long)(kc / P1) * sa_k + (kc % P1);
        float x = Ap[idx];
        if (KG) x = (k < Kd && rowOK) ? x : 0.f;
        f[2 * v + e] = x;
      }
    }
  }
  v16h h;
#pragma unroll
  for (int i = 0; i < 16; ++i) h[i] = (_Float16)f[i];
  return h;
}

template<bool KG, bool CG, bool BK1>
__device__ __forceinline__ v16h loadBfrag(const float* __restrict__ Bp, long colOff,
                                          bool colOK, int kk, int hiB, long sb_k, int Kd)
{
  float f[16];
  if (BK1) {
#pragma unroll
    for (int jj = 0; jj < 4; ++jj) {
      const float4 p = *(const float4*)(Bp + colOff + kk + hiB + 4 * jj); // 16B aligned
      f[4 * jj + 0] = p.x; f[4 * jj + 1] = p.y;
      f[4 * jj + 2] = p.z; f[4 * jj + 3] = p.w;
    }
  } else {
#pragma unroll
    for (int j = 0; j < 16; ++j) {
      const int k  = kk + hiB + j;
      const int kc = KG ? ((k < Kd) ? k : 0) : k;
      float x = Bp[colOff + (long)kc * sb_k];
      if (KG || CG) x = ((!KG || k < Kd) && (!CG || colOK)) ? x : 0.f;
      f[j] = x;
    }
  }
  v16h h;
#pragma unroll
  for (int i = 0; i < 16; ++i) h[i] = (_Float16)f[i];
  return h;
}

// ---------------------------------------------------------------------------
// Generic WMMA GEMM: C[M,Nc] = A[M,Kd] * B[Kd,Nc] (+bias, optional relu)
// One wave per 32x32 C macro-tile: 2x2 fragments, 4 x v_wmma_f32_16x16x32_f16
// per 32-deep k-step. fp32 memory, fp16 compute, fp32 accumulate.
//   A (AMODE 0): idx = (r/ROWDIV)*sa_r1 + (r%ROWDIV)*sa_r0 + k*sa_k
//   A (AMODE 1): idx = (r/P0)*sa_r1 + (k/P1)*sa_k + (r%P0) + (k%P1)  [im2col]
//   B:           idx = k*sb_k + col*sb_c
//   batch z:     z1=z/batchDiv, z0=z%batchDiv; base = z1*Z1 + z0*Z0
// ---------------------------------------------------------------------------
template<int AMODE, int ROWDIV, int P0, int P1,
         bool KG, bool CG, bool AK1, bool BK1, bool BIAS>
__global__ void k_gemm(const float* __restrict__ A, const float* __restrict__ Bm,
                       float* __restrict__ C, const float* __restrict__ bias,
                       int M, int Nc, int Kd,
                       long sa_r1, long sa_r0, long sa_k,
                       long sb_k, long sb_c,
                       int batchDiv,
                       long aZ1, long aZ0, long bZ1, long bZ0, long cZ1, long cZ0,
                       long ldc, int relu)
{
  const int lane = threadIdx.x;
  const int z  = blockIdx.z;
  const int z1 = z / batchDiv, z0 = z - z1 * batchDiv;
  const float* Ap = A  + (long)z1 * aZ1 + (long)z0 * aZ0;
  const float* Bp = Bm + (long)z1 * bZ1 + (long)z0 * bZ0;
  float*       Cp = C  + (long)z1 * cZ1 + (long)z0 * cZ0;

  const int row0 = blockIdx.x * 32, col0 = blockIdx.y * 32;
  const int l15 = lane & 15;
  const int hiA = (lane < 16) ? 0 : 16;
  const int hiB = (lane < 16) ? 0 : 16;

  // two row fragments
  const int  ar0 = row0 + l15, ar1 = ar0 + 16;
  const bool rOK0 = ar0 < M, rOK1 = ar1 < M;
  const int  ar0c = rOK0 ? ar0 : 0, ar1c = rOK1 ? ar1 : 0;
  long rb0, rb1;
  if (AMODE == 0) {
    rb0 = (long)(ar0c / ROWDIV) * sa_r1 + (long)(ar0c % ROWDIV) * sa_r0;
    rb1 = (long)(ar1c / ROWDIV) * sa_r1 + (long)(ar1c % ROWDIV) * sa_r0;
  } else {
    rb0 = (long)(ar0c / P0) * sa_r1 + (ar0c % P0);
    rb1 = (long)(ar1c / P0) * sa_r1 + (ar1c % P0);
  }

  // two col fragments
  const int  bc0 = col0 + l15, bc1 = bc0 + 16;
  const bool cOK0 = bc0 < Nc, cOK1 = bc1 < Nc;
  const long co0 = (long)(cOK0 ? bc0 : 0) * sb_c;
  const long co1 = (long)(cOK1 ? bc1 : 0) * sb_c;

  v8f acc00 = {}, acc01 = {}, acc10 = {}, acc11 = {};
  for (int kk = 0; kk < Kd; kk += 32) {
    const v16h a0 = loadAfrag<AMODE, P1, KG, AK1>(Ap, rb0, rOK0, kk, hiA, sa_k, Kd);
    const v16h a1 = loadAfrag<AMODE, P1, KG, AK1>(Ap, rb1, rOK1, kk, hiA, sa_k, Kd);
    const v16h b0 = loadBfrag<KG, CG, BK1>(Bp, co0, cOK0, kk, hiB, sb_k, Kd);
    const v16h b1 = loadBfrag<KG, CG, BK1>(Bp, co1, cOK1, kk, hiB, sb_k, Kd);
    acc00 = __builtin_amdgcn_wmma_f32_16x16x32_f16(false, a0, false, b0, (short)0, acc00, false, false);
    acc01 = __builtin_amdgcn_wmma_f32_16x16x32_f16(false, a0, false, b1, (short)0, acc01, false, false);
    acc10 = __builtin_amdgcn_wmma_f32_16x16x32_f16(false, a1, false, b0, (short)0, acc10, false, false);
    acc11 = __builtin_amdgcn_wmma_f32_16x16x32_f16(false, a1, false, b1, (short)0, acc11, false, false);
  }

  // C layout: vgpr r -> row = base + r (lanes 0-15) / base + 8 + r (lanes 16-31)
  const int ccA = col0 + l15, ccB = ccA + 16;
  const int rsh = (lane < 16) ? 0 : 8;
  float bia0 = 0.f, bia1 = 0.f;
  if (BIAS) {
    bia0 = bias[(ccA < Nc) ? ccA : 0];
    bia1 = bias[(ccB < Nc) ? ccB : 0];
  }
#pragma unroll
  for (int r = 0; r < 8; ++r) {
    const int row  = row0 + rsh + r;
    const int row2 = row + 16;
    if (row < M) {
      if (ccA < Nc) { float v = acc00[r] + bia0; if (relu) v = fmaxf(v, 0.f); Cp[(long)row * ldc + ccA] = v; }
      if (ccB < Nc) { float v = acc01[r] + bia1; if (relu) v = fmaxf(v, 0.f); Cp[(long)row * ldc + ccB] = v; }
    }
    if (row2 < M) {
      if (ccA < Nc) { float v = acc10[r] + bia0; if (relu) v = fmaxf(v, 0.f); Cp[(long)row2 * ldc + ccA] = v; }
      if (ccB < Nc) { float v = acc11[r] + bia1; if (relu) v = fmaxf(v, 0.f); Cp[(long)row2 * ldc + ccB] = v; }
    }
  }
}

template<int AMODE, int ROWDIV, int P0, int P1,
         bool KG, bool CG, bool AK1, bool BK1, bool BIAS>
static inline void gemm(hipStream_t stream,
                        const float* A, const float* Bm, float* C, const float* bias,
                        int M, int Nc, int Kd,
                        long sa_r1, long sa_r0, long sa_k, long sb_k, long sb_c,
                        int Z, int batchDiv,
                        long aZ1, long aZ0, long bZ1, long bZ0, long cZ1, long cZ0,
                        long ldc, int relu)
{
  dim3 grid((M + 31) / 32, (Nc + 31) / 32, Z);
  k_gemm<AMODE, ROWDIV, P0, P1, KG, CG, AK1, BK1, BIAS><<<grid, 32, 0, stream>>>(
      A, Bm, C, bias, M, Nc, Kd, sa_r1, sa_r0, sa_k, sb_k, sb_c,
      batchDiv, aZ1, aZ0, bZ1, bZ0, cZ1, cZ0, ldc, relu);
}

// ---------------------------------------------------------------------------
// TEm[b,t,n] = LN_n( x[b,n,0,t] + pos_T[t,n] ) * g + b     (512 cols, 256 thr)
// ---------------------------------------------------------------------------
__global__ void k_temporal_embed(const float* __restrict__ x, const float* __restrict__ posT,
                                 const float* __restrict__ g, const float* __restrict__ bta,
                                 float* __restrict__ out)
{
  __shared__ float sh[256];
  const int row = blockIdx.x;             // b*T + t
  const int b = row / T_, t = row % T_;
  const int tid = threadIdx.x;
  const int c0 = tid, c1 = tid + 256;
  const float v0 = x[((long)(b * N_ + c0)) * T_ + t] + posT[t * N_ + c0];
  const float v1 = x[((long)(b * N_ + c1)) * T_ + t] + posT[t * N_ + c1];
  const float s  = blockReduceSum(v0 + v1, sh);
  const float sq = blockReduceSum(v0 * v0 + v1 * v1, sh);
  const float mean = s * (1.f / N_);
  const float var  = sq * (1.f / N_) - mean * mean;
  const float inv  = rsqrtf(var + 1e-5f);
  const long base = (long)row * N_;
  out[base + c0] = (v0 - mean) * inv * g[c0] + bta[c0];
  out[base + c1] = (v1 - mean) * inv * g[c1] + bta[c1];
}

// out[r,:] = LN( in1[r,:] + in2[r % mod, :] ) * g + b   (512 cols, in-place OK)
__global__ void k_add_ln(const float* __restrict__ in1, const float* __restrict__ in2,
                         const float* __restrict__ g, const float* __restrict__ bta,
                         float* __restrict__ out, int mod)
{
  __shared__ float sh[256];
  const int row = blockIdx.x;
  const int tid = threadIdx.x;
  const int c0 = tid, c1 = tid + 256;
  const long base  = (long)row * 512;
  const long base2 = (long)(row % mod) * 512;
  const float v0 = in1[base + c0] + in2[base2 + c0];
  const float v1 = in1[base + c1] + in2[base2 + c1];
  const float s  = blockReduceSum(v0 + v1, sh);
  const float sq = blockReduceSum(v0 * v0 + v1 * v1, sh);
  const float mean = s * (1.f / 512.f);
  const float var  = sq * (1.f / 512.f) - mean * mean;
  const float inv  = rsqrtf(var + 1e-5f);
  out[base + c0] = (v0 - mean) * inv * g[c0] + bta[c0];
  out[base + c1] = (v1 - mean) * inv * g[c1] + bta[c1];
}

// ---------------------------------------------------------------------------
// Temporal attention per (b,h): 12x12 scores (written to d_out tail),
// softmax over the QUERY axis (dim 3, faithful to reference), ctx = attn @ V
// blockDim = 384 (= T*DV)
// ---------------------------------------------------------------------------
__global__ void k_tattn(const float* __restrict__ Qt, const float* __restrict__ Kt,
                        const float* __restrict__ Vt, const float* __restrict__ resPrev,
                        float* __restrict__ scoresOut, float* __restrict__ ctx)
{
  __shared__ float sQ[T_ * 32], sK[T_ * 32], sV[T_ * 32], sS[T_ * T_];
  __shared__ float colMax[T_], colSum[T_];
  const int bh = blockIdx.x;
  const int b = bh / H_, h = bh % H_;
  const int tid = threadIdx.x;
  const int t = tid / 32, d = tid % 32;
  {
    const long rc = (long)(b * T_ + t) * 96 + h * 32 + d;
    sQ[t * 32 + d] = Qt[rc];
    sK[t * 32 + d] = Kt[rc];
    sV[t * 32 + d] = Vt[rc];
  }
  __syncthreads();
  const float invS = 0.17677669529663687f;   // 1/sqrt(32)
  if (tid < T_ * T_) {
    const int q = tid / T_, k = tid % T_;
    float acc = 0.f;
    for (int dd = 0; dd < 32; ++dd) acc += sQ[q * 32 + dd] * sK[k * 32 + dd];
    const long idx = ((long)(b * H_ + h) * T_ + q) * T_ + k;
    const float sc = acc * invS + resPrev[idx];
    scoresOut[idx] = sc;
    sS[q * T_ + k] = sc;
  }
  __syncthreads();
  if (tid < T_) {                           // softmax over q for fixed k
    const int k = tid;
    float mx = -1e30f;
    for (int q = 0; q < T_; ++q) mx = fmaxf(mx, sS[q * T_ + k]);
    float sm = 0.f;
    for (int q = 0; q < T_; ++q) sm += __expf(sS[q * T_ + k] - mx);
    colMax[k] = mx; colSum[k] = sm;
  }
  __syncthreads();
  if (tid < T_ * T_) {
    const int q = tid / T_, k = tid % T_;
    sS[q * T_ + k] = __expf(sS[q * T_ + k] - colMax[k]) / colSum[k];
  }
  __syncthreads();
  {
    float acc = 0.f;
    for (int k = 0; k < T_; ++k) acc += sS[t * T_ + k] * sV[k * 32 + d];
    ctx[(long)(b * T_ + t) * 96 + h * 32 + d] = acc;
  }
}

// ---------------------------------------------------------------------------
// In-place on sat[beta*512+n, m]:
//   v = sat/SCALE + adj[n,m]*mask[kh,n,m];  softmax over m;  *= cheb[kh,n,m]
// ---------------------------------------------------------------------------
__global__ void k_spatial_softmax(float* __restrict__ sat, const float* __restrict__ adj,
                                  const float* __restrict__ maskk, const float* __restrict__ cheb)
{
  __shared__ float sh[256];
  const long r  = blockIdx.x;               // beta*512 + n
  const int  n  = (int)(r & 511);
  const int  kh = (int)((r >> 9) % K_);
  const long base  = r * 512;
  const long mbase = ((long)kh * N_ + n) * N_;
  const long abase = (long)n * N_;
  const int tid = threadIdx.x;
  const int m0 = tid, m1 = tid + 256;
  const float invS = 0.17677669529663687f;
  const float v0 = sat[base + m0] * invS + adj[abase + m0] * maskk[mbase + m0];
  const float v1 = sat[base + m1] * invS + adj[abase + m1] * maskk[mbase + m1];
  const float mx = blockReduceMax(fmaxf(v0, v1), sh);
  const float e0 = __expf(v0 - mx), e1 = __expf(v1 - mx);
  const float sm = blockReduceSum(e0 + e1, sh);
  const float inv = 1.f / sm;
  sat[base + m0] = cheb[mbase + m0] * e0 * inv;
  sat[base + m1] = cheb[mbase + m1] * e1 * inv;
}

// S[bm, o, t] = relu( sum_kh rhs[(b*K+kh), m, t] * Theta[kh, o] )
__global__ void k_spatial_theta(const float* __restrict__ rhs, const float* __restrict__ Theta,
                                float* __restrict__ S, int total)
{
  const int idx = blockIdx.x * blockDim.x + threadIdx.x;
  if (idx >= total) return;
  const int t  = idx % T_;
  const int o  = (idx / T_) % FOUT_;
  const int bm = idx / (T_ * FOUT_);
  const int b = bm / N_, m = bm % N_;
  float acc = 0.f;
#pragma unroll
  for (int kh = 0; kh < K_; ++kh)
    acc += rhs[((long)((b * K_ + kh) * N_ + m)) * T_ + t] * Theta[kh * FOUT_ + o];
  S[idx] = fmaxf(acc, 0.f);
}

// tc[bm, o, colOff+w] = tanh(gt[(bm*W+w), o]) * sigmoid(gt[(bm*W+w), 32+o])
__global__ void k_gtu_act(const float* __restrict__ gt, float* __restrict__ tc,
                          int W, int colOff, int total)
{
  const int idx = blockIdx.x * blockDim.x + threadIdx.x;
  if (idx >= total) return;
  const int w  = idx % W;
  const int o  = (idx / W) % FOUT_;
  const int bm = idx / (W * FOUT_);
  const long r = (long)(bm * W + w) * 64;
  const float a = tanhf(gt[r + o]);
  const float g = 1.f / (1.f + __expf(-gt[r + 32 + o]));
  tc[((long)(bm * FOUT_ + o)) * 24 + colOff + w] = a * g;
}

// out[b,n,o,t] = LN_o( relu( x[b,n,0,t]*res_w[o] + res_b[o] + tcf[bn,o,t] ) )
__global__ void k_out_final(const float* __restrict__ x, const float* __restrict__ resW,
                            const float* __restrict__ resB, const float* __restrict__ tcf,
                            const float* __restrict__ g, const float* __restrict__ bta,
                            float* __restrict__ out, int total)
{
  const int idx = blockIdx.x * blockDim.x + threadIdx.x;
  if (idx >= total) return;
  const int t  = idx % T_;
  const long bn = idx / T_;
  const float xv = x[bn * T_ + t];
  float pre[FOUT_];
  float s = 0.f;
#pragma unroll
  for (int o = 0; o < FOUT_; ++o) {
    float v = xv * resW[o] + resB[o] + tcf[(bn * FOUT_ + o) * T_ + t];
    v = fmaxf(v, 0.f);
    pre[o] = v; s += v;
  }
  const float mean = s * (1.f / FOUT_);
  float sq = 0.f;
#pragma unroll
  for (int o = 0; o < FOUT_; ++o) { const float d = pre[o] - mean; sq += d * d; }
  const float inv = rsqrtf(sq * (1.f / FOUT_) + 1e-5f);
#pragma unroll
  for (int o = 0; o < FOUT_; ++o)
    out[(bn * FOUT_ + o) * T_ + t] = (pre[o] - mean) * inv * g[o] + bta[o];
}

// ---------------------------------------------------------------------------
extern "C" void kernel_launch(void* const* d_in, const int* in_sizes, int n_in,
                              void* d_out, int out_size, void* d_ws, size_t ws_size,
                              hipStream_t stream) {
  (void)in_sizes; (void)n_in; (void)out_size; (void)ws_size;
  const float* x        = (const float*)d_in[0];
  const float* res_prev = (const float*)d_in[1];
  const float* pos_T    = (const float*)d_in[2];
  const float* lnT_g    = (const float*)d_in[3];
  const float* lnT_b    = (const float*)d_in[4];
  const float* Wq_t     = (const float*)d_in[5];
  const float* Wk_t     = (const float*)d_in[6];
  const float* Wv_t     = (const float*)d_in[7];
  const float* fc_t     = (const float*)d_in[8];
  const float* lnA_g    = (const float*)d_in[9];
  const float* lnA_b    = (const float*)d_in[10];
  const float* proj_W   = (const float*)d_in[11];
  const float* proj_b   = (const float*)d_in[12];
  const float* pos_S    = (const float*)d_in[13];
  const float* lnS_g    = (const float*)d_in[14];
  const float* lnS_b    = (const float*)d_in[15];
  const float* Wq_s     = (const float*)d_in[16];
  const float* Wk_s     = (const float*)d_in[17];
  const float* cheb_T   = (const float*)d_in[18];
  const float* adj_pa   = (const float*)d_in[19];
  const float* mask_k   = (const float*)d_in[20];
  const float* Theta    = (const float*)d_in[21];
  const float* gtu3_w   = (const float*)d_in[22];
  const float* gtu3_b   = (const float*)d_in[23];
  const float* gtu5_w   = (const float*)d_in[24];
  const float* gtu5_b   = (const float*)d_in[25];
  const float* gtu7_w   = (const float*)d_in[26];
  const float* gtu7_b   = (const float*)d_in[27];
  const float* fcmy_W   = (const float*)d_in[28];
  const float* fcmy_b   = (const float*)d_in[29];
  const float* res_w    = (const float*)d_in[30];
  const float* res_b    = (const float*)d_in[31];
  const float* lnO_g    = (const float*)d_in[32];
  const float* lnO_b    = (const float*)d_in[33];

  float* outF      = (float*)d_out;
  float* scoresOut = outF + (long)B_ * N_ * FOUT_ * T_;   // 6,291,456

  // workspace bump allocator (fp32 elements)
  float* ws = (float*)d_ws;
  size_t off = 0;
  auto alloc = [&](size_t n) { float* p = ws + off; off += n; return p; };
  float* TEm = alloc(384 * 512);
  float* Qt  = alloc(384 * 96);
  float* Kt  = alloc(384 * 96);
  float* Vt  = alloc(384 * 96);
  float* ctx = alloc(384 * 96);
  float* TAT = alloc(384 * 512);              // ctx@fc_t then LN in-place
  float* xp  = alloc((size_t)16384 * 512);    // proj then SEm in-place
  float* Qs  = alloc((size_t)16384 * 96);
  float* Ks  = alloc((size_t)16384 * 96);
  float* sat = alloc((size_t)96 * 512 * 512); // scores -> T_eff in-place
  float* rhs = alloc((size_t)96 * 512 * 12);
  float* S   = alloc((size_t)16384 * 32 * 12);
  float* tc  = alloc((size_t)524288 * 24);
  float* tcf = alloc((size_t)524288 * 12);
  float* gt  = sat;                           // reuse sat after rhs GEMM done

  // 1) temporal embedding + LN over N
  k_temporal_embed<<<384, 256, 0, stream>>>(x, pos_T, lnT_g, lnT_b, TEm);

  // 2) temporal Q/K/V projections: [384,512] @ [512,96]
  gemm<0,1,1,1, false,false,true,false,false>(stream, TEm, Wq_t, Qt, nullptr, 384, 96, 512,
                      512, 0, 1, 96, 1, 1, 1, 0, 0, 0, 0, 0, 0, 96, 0);
  gemm<0,1,1,1, false,false,true,false,false>(stream, TEm, Wk_t, Kt, nullptr, 384, 96, 512,
                      512, 0, 1, 96, 1, 1, 1, 0, 0, 0, 0, 0, 0, 96, 0);
  gemm<0,1,1,1, false,false,true,false,false>(stream, TEm, Wv_t, Vt, nullptr, 384, 96, 512,
                      512, 0, 1, 96, 1, 1, 1, 0, 0, 0, 0, 0, 0, 96, 0);

  // 3) temporal attention (writes raw scores to d_out tail) + ctx
  k_tattn<<<96, 384, 0, stream>>>(Qt, Kt, Vt, res_prev, scoresOut, ctx);

  // 4) ctx @ fc_t -> TAT buffer; TAT = LN(.. + TEm)
  gemm<0,1,1,1, false,false,true,false,false>(stream, ctx, fc_t, TAT, nullptr, 384, 512, 96,
                      96, 0, 1, 512, 1, 1, 1, 0, 0, 0, 0, 0, 0, 512, 0);
  k_add_ln<<<384, 256, 0, stream>>>(TAT, TEm, lnA_g, lnA_b, TAT, 384);

  // 5) xp = TAT^T-view [16384,12] @ proj_W [12,512] + proj_b ; SEm = LN(xp+pos_S)
  gemm<0,512,1,1, true,false,false,false,true>(stream, TAT, proj_W, xp, proj_b, 16384, 512, 12,
                       6144, 1, 512, 512, 1, 1, 1, 0, 0, 0, 0, 0, 0, 512, 0);
  k_add_ln<<<16384, 256, 0, stream>>>(xp, pos_S, lnS_g, lnS_b, xp, 512);

  // 6) spatial Q/K: [16384,512] @ [512,96]
  gemm<0,1,1,1, false,false,true,false,false>(stream, xp, Wq_s, Qs, nullptr, 16384, 96, 512,
                      512, 0, 1, 96, 1, 1, 1, 0, 0, 0, 0, 0, 0, 96, 0);
  gemm<0,1,1,1, false,false,true,false,false>(stream, xp, Wk_s, Ks, nullptr, 16384, 96, 512,
                      512, 0, 1, 96, 1, 1, 1, 0, 0, 0, 0, 0, 0, 96, 0);

  // 7) sat[beta,n,m] = Qs[b,:,kh] @ Ks[b,:,kh]^T  (96 batched 512x512, Kd=32)
  //    B is k-contiguous (sb_k=1) -> b128 B loads
  gemm<0,1,1,1, false,false,true,true,false>(stream, Qs, Ks, sat, nullptr, 512, 512, 32,
                      96, 0, 1, /*sb_k=*/1, /*sb_c=*/96,
                      96, 3, 512L * 96, 32, 512L * 96, 32,
                      3L * 512 * 512, 512L * 512, 512, 0);

  // 8) fused scale + adj*mask + softmax(m) + cheb multiply (in-place -> T_eff)
  k_spatial_softmax<<<96 * 512, 256, 0, stream>>>(sat, adj_pa, mask_k, cheb_T);

  // 9) rhs[beta,m,t] = T_eff[beta] @ x[b]   (96 batched 512x12, Kd=512)
  gemm<0,1,1,1, false,true,true,false,false>(stream, sat, x, rhs, nullptr, 512, 12, 512,
                     512, 0, 1, /*sb_k=*/12, /*sb_c=*/1,
                     96, 3, 3L * 512 * 512, 512L * 512, 512L * 12, 0,
                     3L * 512 * 12, 512L * 12, 12, 0);

  // 10) Theta contraction + relu -> S[bm, o, t]
  k_spatial_theta<<<(6291456 + 255) / 256, 256, 0, stream>>>(rhs, Theta, S, 6291456);

  // 11) GTU convs as im2col WMMA GEMMs (bias fused, b128 B loads), then gate
  gemm<1,1,10,3, false,false,false,true,true>(stream, S, gtu3_w, gt, gtu3_b, 163840, 64, 96,
                       384, 0, 12, /*sb_k=*/1, /*sb_c=*/96,
                       1, 1, 0, 0, 0, 0, 0, 0, 64, 0);
  k_gtu_act<<<(16384 * 32 * 10 + 255) / 256, 256, 0, stream>>>(gt, tc, 10, 0, 16384 * 32 * 10);

  gemm<1,1,8,5, false,false,false,true,true>(stream, S, gtu5_w, gt, gtu5_b, 131072, 64, 160,
                      384, 0, 12, /*sb_k=*/1, /*sb_c=*/160,
                      1, 1, 0, 0, 0, 0, 0, 0, 64, 0);
  k_gtu_act<<<(16384 * 32 * 8 + 255) / 256, 256, 0, stream>>>(gt, tc, 8, 10, 16384 * 32 * 8);

  gemm<1,1,6,7, false,false,false,true,true>(stream, S, gtu7_w, gt, gtu7_b, 98304, 64, 224,
                      384, 0, 12, /*sb_k=*/1, /*sb_c=*/224,
                      1, 1, 0, 0, 0, 0, 0, 0, 64, 0);
  k_gtu_act<<<(16384 * 32 * 6 + 255) / 256, 256, 0, stream>>>(gt, tc, 6, 18, 16384 * 32 * 6);

  // 12) tc @ fcmy_W + fcmy_b, relu  -> tcf [524288, 12]
  gemm<0,1,1,1, true,true,false,false,true>(stream, tc, fcmy_W, tcf, fcmy_b, 524288, 12, 24,
                     24, 0, 1, /*sb_k=*/12, /*sb_c=*/1,
                     1, 1, 0, 0, 0, 0, 0, 0, 12, 1);

  // 13) residual + relu + LN over Fout -> out [B,N,Fout,T]
  k_out_final<<<768, 256, 0, stream>>>(x, res_w, res_b, tcf, lnO_g, lnO_b, outF,
                                       B_ * N_ * T_);
}
